// MyValModel_25890062860837
// MI455X (gfx1250) — compile-verified, matrix operand order checked by reading, not run on hardware
//
#include <hip/hip_runtime.h>
#include <math.h>

// ---------------------------------------------------------------------------
// Model constants (from the reference)
// ---------------------------------------------------------------------------
constexpr int BATCH  = 8;
constexpr int NFEAT  = 64;
constexpr int NHID   = 128;
constexpr int NCLS   = 100;
constexpr int N_SU   = 2076;
constexpr int N_SV   = 16335;
constexpr int DEG    = 6;
constexpr int NMAX   = BATCH * N_SV;          // 130680 nodes (solvent, the larger graph)
constexpr int EMAX   = BATCH * N_SV * DEG;    // 784080 edges
constexpr int LDO    = 128;                   // uniform row stride for node buffers

// CDNA5 WMMA vector types (probe-verified signatures)
typedef __attribute__((ext_vector_type(16))) __bf16 v16bf;
typedef __attribute__((ext_vector_type(8)))  float  v8f;

__device__ __forceinline__ __bf16 f2bf(float f) { return (__bf16)f; }
__device__ __forceinline__ float sigm(float x) { return 1.0f / (1.0f + __expf(-x)); }

// Order-preserving float<->uint encoding for atomic max over signed floats
__device__ __forceinline__ unsigned enc_f(float x) {
    unsigned u = __float_as_uint(x);
    return (u & 0x80000000u) ? ~u : (u | 0x80000000u);
}
__device__ __forceinline__ float dec_f(unsigned e) {
    unsigned u = (e & 0x80000000u) ? (e ^ 0x80000000u) : ~e;
    return __uint_as_float(u);
}
// enc(-inf) == ~0xFF800000 == 0x007FFFFF
#define ENC_NEG_INF 0x007FFFFFu

// Pack 8 fp32 (two float4) into bf16 elements [base, base+8) of a v16bf.
__device__ __forceinline__ void pack8(v16bf& V, int base, float4 u, float4 v) {
    V[base + 0] = f2bf(u.x); V[base + 1] = f2bf(u.y);
    V[base + 2] = f2bf(u.z); V[base + 3] = f2bf(u.w);
    V[base + 4] = f2bf(v.x); V[base + 5] = f2bf(v.y);
    V[base + 6] = f2bf(v.z); V[base + 7] = f2bf(v.w);
}

// ---------------------------------------------------------------------------
// WMMA GEMM:  Out[N, Dout] = X[N, Din] @ W[Dout, Din]^T + bias   (opt. ReLU)
// One wave computes one 16x16 tile; 8 waves per block cover 128 rows.
// DIN is a template constant (64 or 128) so the K-loop fully unrolls into
// back-to-back v_wmma_f32_16x16x32_bf16 with batched global_load_b128s.
//
// OOB handling: A rows past Nrows / B rows past Dout are *clamped* to a valid
// row (not zero-filled) — the garbage D rows/cols are masked at the store, so
// loads stay unconditional vector loads (no per-element EXEC gymnastics).
// Interior tiles (the overwhelmingly common case; solute N is an exact
// multiple of 16) take an unguarded store path: 8 coalesced b32 stores via a
// strength-reduced base pointer, no per-row exec juggling.
//
// Per-lane operand layout per CDNA5 ISA (16-bit A 16x32):
//   lane = 16*half + r ;  element e<8  -> K = kk + 8*half + e
//                         element e>=8 -> K = kk + 16 + 8*half + (e-8)
// B uses the identical mapping with r indexing the output dim (W row).
// D:  col = n0 + (lane&15), row = m0 + v + 8*half  for acc element v.
// ---------------------------------------------------------------------------
template <int DIN>
__global__ __launch_bounds__(256)
void wmma_gemm_xWt(const float* __restrict__ X, int ldx,
                   const float* __restrict__ W, const float* __restrict__ bias,
                   int Dout, float* __restrict__ Out, int ldo,
                   int Nrows, int relu)
{
    const int lane  = threadIdx.x & 31;
    const int wave  = threadIdx.x >> 5;
    const int mtile = blockIdx.y * 8 + wave;
    const int m0    = mtile * 16;
    const int n0    = blockIdx.x * 16;
    if (m0 >= Nrows) return;                 // wave-uniform: EXEC stays full for WMMA

    const int half = lane >> 4;
    const int r15  = lane & 15;

    int arow = m0 + r15;  if (arow >= Nrows) arow = Nrows - 1;   // clamp, not zero
    int brow = n0 + r15;  if (brow >= Dout)  brow = Dout - 1;
    const float* __restrict__ ap = X + (size_t)arow * ldx + 8 * half;
    const float* __restrict__ bp = W + (size_t)brow * DIN + 8 * half;

    v8f acc = {};
#pragma unroll
    for (int kk = 0; kk < DIN; kk += 32) {
        __builtin_prefetch(ap + kk + 32, 0, 1);     // global_prefetch_b8
        // Two 8-wide K runs per operand, each as 2 aligned float4 loads.
        const float4 a0 = *(const float4*)(ap + kk);
        const float4 a1 = *(const float4*)(ap + kk + 4);
        const float4 a2 = *(const float4*)(ap + kk + 16);
        const float4 a3 = *(const float4*)(ap + kk + 20);
        const float4 b0 = *(const float4*)(bp + kk);
        const float4 b1 = *(const float4*)(bp + kk + 4);
        const float4 b2 = *(const float4*)(bp + kk + 16);
        const float4 b3 = *(const float4*)(bp + kk + 20);
        v16bf A, Bm;
        pack8(A,  0, a0, a1);  pack8(A,  8, a2, a3);
        pack8(Bm, 0, b0, b1);  pack8(Bm, 8, b2, b3);
        acc = __builtin_amdgcn_wmma_f32_16x16x32_bf16(
                  /*neg_a=*/false, A, /*neg_b=*/false, Bm,
                  /*c_mod=*/(short)0, acc, /*reuse_a=*/false, /*reuse_b=*/false);
    }

    const int col = n0 + r15;
    if (col >= Dout) return;
    const float bb = bias[col];
    float* __restrict__ op = Out + (size_t)(m0 + 8 * half) * ldo + col;

    if (m0 + 15 < Nrows) {
        // Interior tile: unconditional coalesced stores, pointer-stepped.
#pragma unroll
        for (int v = 0; v < 8; ++v) {
            float o = acc[v] + bb;
            if (relu) o = fmaxf(o, 0.0f);
            op[(size_t)v * ldo] = o;
        }
    } else {
        // Ragged final tile only.
#pragma unroll
        for (int v = 0; v < 8; ++v) {
            const int row = m0 + v + 8 * half;
            if (row < Nrows) {
                float o = acc[v] + bb;
                if (relu) o = fmaxf(o, 0.0f);
                op[(size_t)v * ldo] = o;
            }
        }
    }
}

// ---------------------------------------------------------------------------
// Edge attention (TransformerConv, heads=1), 3-pass atomic formulation.
// numer/den/amax fit in MI455X's 192MB L2, so atomics resolve on-chip.
// ---------------------------------------------------------------------------
__global__ void init_node_bufs(float* __restrict__ numer, float* __restrict__ den,
                               unsigned* __restrict__ amax, int Nn)
{
    const int i = blockIdx.x * 256 + threadIdx.x;
    if (i < Nn) { den[i] = 0.0f; amax[i] = ENC_NEG_INF; }
    if (i < Nn * LDO) numer[i] = 0.0f;
}

__global__ __launch_bounds__(256)
void edge_attn_pass1(const int* __restrict__ src, const int* __restrict__ dst,
                     const float* __restrict__ Q, const float* __restrict__ K,
                     int dout, int E, float scale,
                     float* __restrict__ abuf, unsigned* __restrict__ amax)
{
    const int e = blockIdx.x * 8 + (threadIdx.x >> 5);
    if (e >= E) return;
    const int lane = threadIdx.x & 31;
    const int s = src[e], d = dst[e];
    const float* __restrict__ q = Q + (size_t)d * LDO;
    const float* __restrict__ k = K + (size_t)s * LDO;
    float sum = 0.0f;
    for (int j = lane; j < dout; j += 32) sum += q[j] * k[j];
    for (int off = 16; off; off >>= 1) sum += __shfl_xor(sum, off, 32);
    if (lane == 0) {
        const float a = sum * scale;
        abuf[e] = a;
        atomicMax(amax + d, enc_f(a));
    }
}

__global__ __launch_bounds__(256)
void edge_attn_pass2(const int* __restrict__ src, const int* __restrict__ dst,
                     const float* __restrict__ V, int dout, int E,
                     const float* __restrict__ abuf, const unsigned* __restrict__ amax,
                     float* __restrict__ numer, float* __restrict__ den)
{
    const int e = blockIdx.x * 8 + (threadIdx.x >> 5);
    if (e >= E) return;
    const int lane = threadIdx.x & 31;
    const int s = src[e], d = dst[e];
    const float ae = __expf(abuf[e] - dec_f(amax[d]));
    const float* __restrict__ v = V + (size_t)s * LDO;
    float* __restrict__ nu = numer + (size_t)d * LDO;
    for (int j = lane; j < dout; j += 32) atomicAdd(nu + j, ae * v[j]);
    if (lane == 0) atomicAdd(den + d, ae);
}

__global__ void attn_finalize(const float* __restrict__ numer, const float* __restrict__ den,
                              const float* __restrict__ skipb,
                              const float* __restrict__ initb,   // may be null
                              int dout, int Nn, float* __restrict__ out, int relu)
{
    const int n = blockIdx.x;
    if (n >= Nn) return;
    const float inv = 1.0f / (den[n] + 1e-16f);
    for (int j = threadIdx.x; j < dout; j += blockDim.x) {
        float o = numer[(size_t)n * LDO + j] * inv + skipb[(size_t)n * LDO + j];
        if (initb) o += initb[(size_t)n * LDO + j];
        if (relu) o = fmaxf(o, 0.0f);
        out[(size_t)n * LDO + j] = o;
    }
}

// ---------------------------------------------------------------------------
// SMILES branch: per-token GRU step with h0=0 (tokens independent), relu,
// mean over 100 tokens.  block = (seq in {0,1}, dir in {0,1}, token).
// ---------------------------------------------------------------------------
__global__ __launch_bounds__(128)
void smiles_gru(const int* __restrict__ smiles, const float* __restrict__ embed,
                const float* __restrict__ wih_f, const float* __restrict__ bih_f,
                const float* __restrict__ bhh_f,
                const float* __restrict__ wih_b, const float* __restrict__ bih_b,
                const float* __restrict__ bhh_b,
                float* __restrict__ seqvec /* [2][200], pre-zeroed */)
{
    const int bid  = blockIdx.x;
    const int l    = bid % 100;
    const int dir  = (bid / 100) & 1;
    const int sidx = bid / 200;                      // 0 -> smiles[0], 1 -> smiles[5]
    const int tok  = smiles[(sidx == 0 ? 0 : 5) * 100 + l];
    const float* __restrict__ x   = embed + tok * 100;
    const float* __restrict__ wih = dir ? wih_b : wih_f;
    const float* __restrict__ bih = dir ? bih_b : bih_f;
    const float* __restrict__ bhh = dir ? bhh_b : bhh_f;
    const int t = threadIdx.x;
    if (t >= 100) return;
    float gr = bih[t], gz = bih[100 + t], gn = bih[200 + t];
    for (int j = 0; j < 100; ++j) {
        const float xv = x[j];
        gr += wih[t * 100 + j] * xv;
        gz += wih[(100 + t) * 100 + j] * xv;
        gn += wih[(200 + t) * 100 + j] * xv;
    }
    const float r = sigm(gr + bhh[t]);
    const float z = sigm(gz + bhh[100 + t]);
    const float n = tanhf(gn + r * bhh[200 + t]);
    const float h = (1.0f - z) * n;                  // h0 == 0
    atomicAdd(&seqvec[sidx * 200 + dir * 100 + t], fmaxf(h, 0.0f) * 0.01f);
}

// ---------------------------------------------------------------------------
// set2set (steps=2, H=100)
// ---------------------------------------------------------------------------
__global__ void zero_buf(float* __restrict__ p, int n)
{
    const int i = blockIdx.x * 256 + threadIdx.x;
    if (i < n) p[i] = 0.0f;
}

__global__ __launch_bounds__(128)
void s2s_gates(const float* __restrict__ wih, const float* __restrict__ whh,
               const float* __restrict__ bih, const float* __restrict__ bhh,
               float* __restrict__ h, float* __restrict__ c,
               const float* __restrict__ qstar)
{
    __shared__ float g[400];
    const int b = blockIdx.x;
    const float* __restrict__ q  = qstar + b * 200;
    const float* __restrict__ hb = h + b * 100;
    for (int row = threadIdx.x; row < 400; row += 128) {
        float s = bih[row] + bhh[row];
        const float* __restrict__ wi = wih + row * 200;
        for (int j = 0; j < 200; ++j) s += wi[j] * q[j];
        const float* __restrict__ wh = whh + row * 100;
        for (int j = 0; j < 100; ++j) s += wh[j] * hb[j];
        g[row] = s;
    }
    __syncthreads();
    const int t = threadIdx.x;
    if (t < 100) {                                   // gate order i,f,g,o
        const float cn = sigm(g[100 + t]) * c[b * 100 + t] + sigm(g[t]) * tanhf(g[200 + t]);
        c[b * 100 + t] = cn;
        h[b * 100 + t] = sigm(g[300 + t]) * tanhf(cn);
    }
}

__global__ __launch_bounds__(256)
void s2s_energy(const float* __restrict__ x, int npg, const float* __restrict__ h,
                float* __restrict__ e)
{
    const int node = blockIdx.x * 8 + (threadIdx.x >> 5);
    if (node >= BATCH * npg) return;
    const int b = node / npg;
    const int lane = threadIdx.x & 31;
    const float* __restrict__ xr = x + (size_t)node * LDO;
    const float* __restrict__ hb = h + b * 100;
    float s = 0.0f;
    for (int j = lane; j < 100; j += 32) s += xr[j] * hb[j];
    for (int off = 16; off; off >>= 1) s += __shfl_xor(s, off, 32);
    if (lane == 0) e[node] = s;
}

__global__ __launch_bounds__(256)
void s2s_reduce(const float* __restrict__ e, int npg,
                float* __restrict__ emax, float* __restrict__ esum, float* __restrict__ r)
{
    __shared__ float sm[256];
    const int b = blockIdx.x;
    const float* __restrict__ eb = e + (size_t)b * npg;
    float m = -INFINITY;
    for (int i = threadIdx.x; i < npg; i += 256) m = fmaxf(m, eb[i]);
    sm[threadIdx.x] = m; __syncthreads();
    for (int s = 128; s; s >>= 1) { if (threadIdx.x < s) sm[threadIdx.x] = fmaxf(sm[threadIdx.x], sm[threadIdx.x + s]); __syncthreads(); }
    const float mx = sm[0]; __syncthreads();
    float su = 0.0f;
    for (int i = threadIdx.x; i < npg; i += 256) su += __expf(eb[i] - mx);
    sm[threadIdx.x] = su; __syncthreads();
    for (int s = 128; s; s >>= 1) { if (threadIdx.x < s) sm[threadIdx.x] += sm[threadIdx.x + s]; __syncthreads(); }
    if (threadIdx.x == 0) { emax[b] = mx; esum[b] = sm[0]; }
    if (threadIdx.x < 100) r[b * 100 + threadIdx.x] = 0.0f;   // zero r before readout
}

__global__ __launch_bounds__(256)
void s2s_readout(const float* __restrict__ x, int npg, const float* __restrict__ e,
                 const float* __restrict__ emax, const float* __restrict__ esum,
                 float* __restrict__ r)
{
    const int node = blockIdx.x * 8 + (threadIdx.x >> 5);
    if (node >= BATCH * npg) return;
    const int b = node / npg;
    const int lane = threadIdx.x & 31;
    const float w = __expf(e[node] - emax[b]) / esum[b];
    const float* __restrict__ xr = x + (size_t)node * LDO;
    float* __restrict__ rb = r + b * 100;
    for (int j = lane; j < 100; j += 32) atomicAdd(rb + j, w * xr[j]);
}

__global__ void s2s_qstar(const float* __restrict__ h, const float* __restrict__ r,
                          float* __restrict__ qstar)
{
    const int b = blockIdx.x, t = threadIdx.x;
    if (t < 100)      qstar[b * 200 + t] = h[b * 100 + t];
    else if (t < 200) qstar[b * 200 + t] = r[b * 100 + (t - 100)];
}

// ---------------------------------------------------------------------------
// MLP head: concat([su, after_su, sv, after_sv]) -> fc2..fc5.  Block per batch.
// ---------------------------------------------------------------------------
__global__ __launch_bounds__(256)
void head_mlp(const float* __restrict__ su_q, const float* __restrict__ sv_q,
              const float* __restrict__ seqvec,
              const float* __restrict__ fc2w, const float* __restrict__ fc2b,
              const float* __restrict__ fc3w, const float* __restrict__ fc3b,
              const float* __restrict__ fc4w, const float* __restrict__ fc4b,
              const float* __restrict__ fc5w, const float* __restrict__ fc5b,
              float* __restrict__ out)
{
    __shared__ float data[800];
    __shared__ float h1[100];
    __shared__ float h2[64];
    __shared__ float h3[32];
    const int b = blockIdx.x, t = threadIdx.x;
    for (int j = t; j < 200; j += 256) {
        data[j]       = su_q[b * 200 + j];
        data[200 + j] = seqvec[j];         // after_su (tiled)
        data[400 + j] = sv_q[b * 200 + j];
        data[600 + j] = seqvec[200 + j];   // after_sv
    }
    __syncthreads();
    if (t < 100) { float s = fc2b[t]; const float* w = fc2w + t * 800; for (int j = 0; j < 800; ++j) s += w[j] * data[j]; h1[t] = fmaxf(s, 0.0f); }
    __syncthreads();
    if (t < 64)  { float s = fc3b[t]; const float* w = fc3w + t * 100; for (int j = 0; j < 100; ++j) s += w[j] * h1[j];  h2[t] = fmaxf(s, 0.0f); }
    __syncthreads();
    if (t < 32)  { float s = fc4b[t]; const float* w = fc4w + t * 64;  for (int j = 0; j < 64;  ++j) s += w[j] * h2[j];  h3[t] = fmaxf(s, 0.0f); }
    __syncthreads();
    if (t == 0)  { float s = fc5b[0]; for (int j = 0; j < 32; ++j) s += fc5w[j] * h3[j]; out[b] = s; }
}

// ---------------------------------------------------------------------------
// Host orchestration.  Input order = setup_inputs() insertion order, with
// params flattened recursively in dict-insertion order (42 leaves total).
// ---------------------------------------------------------------------------
extern "C" void kernel_launch(void* const* d_in, const int* in_sizes, int n_in,
                              void* d_out, int out_size, void* d_ws, size_t ws_size,
                              hipStream_t stream)
{
    (void)in_sizes; (void)n_in; (void)out_size; (void)ws_size;

    const int*   solute_adj  = (const int*)  d_in[0];
    const float* solute_x    = (const float*)d_in[1];
    const float* solvent_x   = (const float*)d_in[2];
    const int*   solvent_adj = (const int*)  d_in[3];
    const int*   smiles      = (const int*)  d_in[4];

    int pi = 5;
    auto F = [&](void) { return (const float*)d_in[pi++]; };
    const float* p_embed     = F();
    const float* p_gwih_f = F(); const float* p_gbih_f = F(); const float* p_gbhh_f = F();
    const float* p_gwih_b = F(); const float* p_gbih_b = F(); const float* p_gbhh_b = F();
    const float* p_fc1w = F();   const float* p_fc1b = F();
    const float* c1_wq = F(); const float* c1_bq = F();
    const float* c1_wk = F(); const float* c1_bk = F();
    const float* c1_wv = F(); const float* c1_bv = F();
    const float* c1_ws = F(); const float* c1_bs = F();
    const float* c2_wq = F(); const float* c2_bq = F();
    const float* c2_wk = F(); const float* c2_bk = F();
    const float* c2_wv = F(); const float* c2_bv = F();
    const float* c2_ws = F(); const float* c2_bs = F();
    const float* p_lwih = F(); const float* p_lwhh = F();
    const float* p_lbih = F(); const float* p_lbhh = F();
    const float* p_fc2w = F(); const float* p_fc2b = F();
    const float* p_fc3w = F(); const float* p_fc3b = F();
    const float* p_fc4w = F(); const float* p_fc4b = F();
    const float* p_fc5w = F(); const float* p_fc5b = F();

    // Workspace carve-up (floats); branches run sequentially and share buffers.
    float* w = (float*)d_ws;
    size_t off = 0;
    auto alloc = [&](size_t n) { float* p = w + off; off += n; return p; };
    float*    buf_q     = alloc((size_t)NMAX * LDO);
    float*    buf_k     = alloc((size_t)NMAX * LDO);
    float*    buf_v     = alloc((size_t)NMAX * LDO);
    float*    buf_skip  = alloc((size_t)NMAX * LDO);
    float*    buf_x     = alloc((size_t)NMAX * LDO);   // conv1 out / conv2 in
    float*    buf_numer = alloc((size_t)NMAX * LDO);
    float*    buf_init  = alloc((size_t)NMAX * LDO);
    float*    buf_x2    = alloc((size_t)NMAX * LDO);   // conv2 out (set2set input)
    float*    buf_den   = alloc(NMAX);
    unsigned* buf_amax  = (unsigned*)alloc(NMAX);
    float*    buf_a     = alloc(EMAX);
    float*    buf_e     = alloc(NMAX);
    float*    su_q      = alloc(BATCH * 200);
    float*    sv_q      = alloc(BATCH * 200);
    float*    s_h       = alloc(BATCH * 100);
    float*    s_c       = alloc(BATCH * 100);
    float*    s_r       = alloc(BATCH * 100);
    float*    s_emax    = alloc(BATCH);
    float*    s_esum    = alloc(BATCH);
    float*    seqvec    = alloc(2 * 200);

    // --- SMILES branch (independent; run first) ---
    zero_buf<<<2, 256, 0, stream>>>(seqvec, 400);
    smiles_gru<<<400, 128, 0, stream>>>(smiles, p_embed,
                                        p_gwih_f, p_gbih_f, p_gbhh_f,
                                        p_gwih_b, p_gbih_b, p_gbhh_b, seqvec);

    // --- Graph branch (shared scratch; sequential on stream) ---
    auto run_branch = [&](const float* feat, const int* adj, int npg, float* qstar_out) {
        const int Nn = BATCH * npg;
        const int E  = Nn * DEG;
        const int* src = adj;        // ei[0]
        const int* dst = adj + E;    // ei[1]

        auto gemm64 = [&](const float* X, int ldx, const float* W, const float* bias,
                          int Dout, float* Out, int relu) {
            dim3 grid((Dout + 15) / 16, (Nn + 127) / 128);
            wmma_gemm_xWt<64><<<grid, 256, 0, stream>>>(X, ldx, W, bias, Dout, Out, LDO, Nn, relu);
        };
        auto gemm128 = [&](const float* X, int ldx, const float* W, const float* bias,
                           int Dout, float* Out, int relu) {
            dim3 grid((Dout + 15) / 16, (Nn + 127) / 128);
            wmma_gemm_xWt<128><<<grid, 256, 0, stream>>>(X, ldx, W, bias, Dout, Out, LDO, Nn, relu);
        };

        // init = fc1(feat) (no relu) ; conv1 projections (Din = 64)
        gemm64(feat, NFEAT, p_fc1w, p_fc1b, NCLS, buf_init, 0);
        gemm64(feat, NFEAT, c1_wq, c1_bq, NHID, buf_q, 0);
        gemm64(feat, NFEAT, c1_wk, c1_bk, NHID, buf_k, 0);
        gemm64(feat, NFEAT, c1_wv, c1_bv, NHID, buf_v, 0);
        gemm64(feat, NFEAT, c1_ws, c1_bs, NHID, buf_skip, 0);

        const int gE = (E + 7) / 8;
        init_node_bufs<<<(Nn * LDO + 255) / 256, 256, 0, stream>>>(buf_numer, buf_den, buf_amax, Nn);
        edge_attn_pass1<<<gE, 256, 0, stream>>>(src, dst, buf_q, buf_k, NHID, E,
                                                1.0f / sqrtf((float)NHID), buf_a, buf_amax);
        edge_attn_pass2<<<gE, 256, 0, stream>>>(src, dst, buf_v, NHID, E, buf_a, buf_amax,
                                                buf_numer, buf_den);
        attn_finalize<<<Nn, 128, 0, stream>>>(buf_numer, buf_den, buf_skip, nullptr,
                                              NHID, Nn, buf_x, /*relu=*/1);

        // conv2 (Din = 128, dout = 100) + init skip, no relu
        gemm128(buf_x, LDO, c2_wq, c2_bq, NCLS, buf_q, 0);
        gemm128(buf_x, LDO, c2_wk, c2_bk, NCLS, buf_k, 0);
        gemm128(buf_x, LDO, c2_wv, c2_bv, NCLS, buf_v, 0);
        gemm128(buf_x, LDO, c2_ws, c2_bs, NCLS, buf_skip, 0);

        init_node_bufs<<<(Nn * LDO + 255) / 256, 256, 0, stream>>>(buf_numer, buf_den, buf_amax, Nn);
        edge_attn_pass1<<<gE, 256, 0, stream>>>(src, dst, buf_q, buf_k, NCLS, E,
                                                1.0f / sqrtf((float)NCLS), buf_a, buf_amax);
        edge_attn_pass2<<<gE, 256, 0, stream>>>(src, dst, buf_v, NCLS, E, buf_a, buf_amax,
                                                buf_numer, buf_den);
        attn_finalize<<<Nn, 128, 0, stream>>>(buf_numer, buf_den, buf_skip, buf_init,
                                              NCLS, Nn, buf_x2, /*relu=*/0);

        // set2set, steps = 2
        zero_buf<<<(BATCH * 100 + 255) / 256, 256, 0, stream>>>(s_h, BATCH * 100);
        zero_buf<<<(BATCH * 100 + 255) / 256, 256, 0, stream>>>(s_c, BATCH * 100);
        zero_buf<<<(BATCH * 200 + 255) / 256, 256, 0, stream>>>(qstar_out, BATCH * 200);
        const int gN = (Nn + 7) / 8;
        for (int step = 0; step < 2; ++step) {
            s2s_gates<<<BATCH, 128, 0, stream>>>(p_lwih, p_lwhh, p_lbih, p_lbhh, s_h, s_c, qstar_out);
            s2s_energy<<<gN, 256, 0, stream>>>(buf_x2, npg, s_h, buf_e);
            s2s_reduce<<<BATCH, 256, 0, stream>>>(buf_e, npg, s_emax, s_esum, s_r);
            s2s_readout<<<gN, 256, 0, stream>>>(buf_x2, npg, buf_e, s_emax, s_esum, s_r);
            s2s_qstar<<<BATCH, 256, 0, stream>>>(s_h, s_r, qstar_out);
        }
    };

    run_branch(solute_x,  solute_adj,  N_SU, su_q);
    run_branch(solvent_x, solvent_adj, N_SV, sv_q);

    head_mlp<<<BATCH, 256, 0, stream>>>(su_q, sv_q, seqvec,
                                        p_fc2w, p_fc2b, p_fc3w, p_fc3b,
                                        p_fc4w, p_fc4b, p_fc5w, p_fc5b,
                                        (float*)d_out);
}
